// PatchQuantumGenerator_27711128994129
// MI455X (gfx1250) — compile-verified, matrix operand order checked by reading, not run on hardware
//
#include <hip/hip_runtime.h>
#include <hip/hip_bf16.h>

// ---------------------------------------------------------------------------
// Problem constants
// ---------------------------------------------------------------------------
#define NQ      6
#define QDEPTH  6
#define NGEN    4
#define DIM     64          // 2^NQ
#define IMG     256         // NGEN * DIM
#define BATCH   8192
#define HPAD    512         // 500 padded to 512

typedef __attribute__((ext_vector_type(16))) _Float16 v16h;
typedef __attribute__((ext_vector_type(8)))  _Float16 v8h;
typedef __attribute__((ext_vector_type(8)))  float    v8f;

union FragH { v16h v; v8h h[2]; };

// ---------------------------------------------------------------------------
// Quantum state-vector simulation: one wave32 per (generator, sample).
// 64 amplitudes -> 2 complex amplitudes per lane (a = 2*lane + j).
// Qubit at bit p: p==0 pairs are intra-lane; p>=1 partner via shfl_xor(1<<(p-1)).
// ---------------------------------------------------------------------------
__device__ __forceinline__ void apply_ry(float theta, int p, int lane,
                                         float& re0, float& im0,
                                         float& re1, float& im1) {
    float s, c;
    __sincosf(theta * 0.5f, &s, &c);
    if (p == 0) {
        // local pair (a0=2L, a1=2L+1)
        float nr0 = c * re0 - s * re1;
        float ni0 = c * im0 - s * im1;
        float nr1 = s * re0 + c * re1;
        float ni1 = s * im0 + c * im1;
        re0 = nr0; im0 = ni0; re1 = nr1; im1 = ni1;
    } else {
        int mask = 1 << (p - 1);
        float pre0 = __shfl_xor(re0, mask, 32);
        float pim0 = __shfl_xor(im0, mask, 32);
        float pre1 = __shfl_xor(re1, mask, 32);
        float pim1 = __shfl_xor(im1, mask, 32);
        float sgn = ((lane >> (p - 1)) & 1) ? s : -s;
        re0 = c * re0 + sgn * pre0;  im0 = c * im0 + sgn * pim0;
        re1 = c * re1 + sgn * pre1;  im1 = c * im1 + sgn * pim1;
    }
}

__device__ __forceinline__ void apply_rx(float theta, int p, int lane,
                                         float& re0, float& im0,
                                         float& re1, float& im1) {
    float s, c;
    __sincosf(theta * 0.5f, &s, &c);
    // new = c*cur - i*s*partner  (row-symmetric)
    if (p == 0) {
        float nr0 = c * re0 + s * im1;
        float ni0 = c * im0 - s * re1;
        float nr1 = c * re1 + s * im0;
        float ni1 = c * im1 - s * re0;
        re0 = nr0; im0 = ni0; re1 = nr1; im1 = ni1;
    } else {
        int mask = 1 << (p - 1);
        float pre0 = __shfl_xor(re0, mask, 32);
        float pim0 = __shfl_xor(im0, mask, 32);
        float pre1 = __shfl_xor(re1, mask, 32);
        float pim1 = __shfl_xor(im1, mask, 32);
        re0 = c * re0 + s * pim0;  im0 = c * im0 - s * pre0;
        re1 = c * re1 + s * pim1;  im1 = c * im1 - s * pre1;
    }
    (void)lane;
}

__global__ __launch_bounds__(256)
void qsim_kernel(const float* __restrict__ x,      // (B, 2*NQ)
                 const float* __restrict__ qp,     // (NGEN, QDEPTH*NQ*2)
                 _Float16* __restrict__ feat)      // (B, IMG) f16
{
    const int lane = threadIdx.x & 31;
    const int wid  = blockIdx.x * (blockDim.x >> 5) + (threadIdx.x >> 5);
    const int g = wid >> 13;        // / BATCH
    const int b = wid & (BATCH - 1);

    const float* noise = x + b * (2 * NQ);
    const float* w     = qp + g * (QDEPTH * NQ * 2);

    // amplitudes a0 = 2*lane, a1 = 2*lane+1
    float re0 = (lane == 0) ? 1.0f : 0.0f, im0 = 0.0f;
    float re1 = 0.0f, im1 = 0.0f;

    // noise embedding: wire i -> bit p = NQ-1-i (wire 0 is MSB)
    #pragma unroll
    for (int i = 0; i < NQ; ++i) {
        const int p = NQ - 1 - i;
        apply_ry(noise[i],      p, lane, re0, im0, re1, im1);
        apply_rx(noise[i + NQ], p, lane, re0, im0, re1, im1);
    }

    const unsigned a0 = 2u * lane, a1 = a0 + 1u;
    for (int d = 0; d < QDEPTH; ++d) {
        #pragma unroll
        for (int y = 0; y < NQ; ++y) {
            const int p = NQ - 1 - y;
            apply_rx(w[(d * NQ + y) * 2 + 0], p, lane, re0, im0, re1, im1);
            apply_ry(w[(d * NQ + y) * 2 + 1], p, lane, re0, im0, re1, im1);
        }
        // CZ chain: negate amplitudes where bits (NQ-1-y) and (NQ-2-y) both set
        #pragma unroll
        for (int y = 0; y < NQ - 1; ++y) {
            const int p1 = NQ - 1 - y, p2 = NQ - 2 - y;
            if ((((a0 >> p1) & 1u) & ((a0 >> p2) & 1u)) != 0u) { re0 = -re0; im0 = -im0; }
            if ((((a1 >> p1) & 1u) & ((a1 >> p2) & 1u)) != 0u) { re1 = -re1; im1 = -im1; }
        }
    }

    const float p0 = re0 * re0 + im0 * im0;
    const float p1 = re1 * re1 + im1 * im1;
    _Float16* o = feat + (size_t)b * IMG + g * DIM + 2 * lane;
    o[0] = (_Float16)p0;
    o[1] = (_Float16)p1;
}

// ---------------------------------------------------------------------------
// Weight prep: transpose + pad + f32->f16.  Wt[n*Kpad + k] = W[k*N + n]
// ---------------------------------------------------------------------------
__global__ void transpose_pad_f16(const float* __restrict__ W,
                                  _Float16* __restrict__ Wt,
                                  int K, int N, int Kpad, int Npad) {
    int idx = blockIdx.x * blockDim.x + threadIdx.x;
    if (idx >= Kpad * Npad) return;
    int k = idx % Kpad;
    int n = idx / Kpad;
    float v = (k < K && n < N) ? W[k * N + n] : 0.0f;
    Wt[(size_t)n * Kpad + k] = (_Float16)v;
}

__global__ void pad_bias(const float* __restrict__ b, float* __restrict__ bp,
                         int N, int Npad) {
    int i = blockIdx.x * blockDim.x + threadIdx.x;
    if (i < Npad) bp[i] = (i < N) ? b[i] : 0.0f;
}

// ---------------------------------------------------------------------------
// WMMA GEMM:  C(M x Npad) = act( A(M x Kpad) @ Wt^T + bias )
//   A  : f16 row-major, stride Kpad
//   Wt : f16 (Npad x Kpad) -> lane-contiguous B fragments
//   256 threads = 8 waves; block tile 256(M) x 64(N); each wave: 32(M) x 64(N)
//   (two M-subtiles per wave -> B fragments reused across both, 12 b128 loads
//    feed 8 WMMAs per K-step).
//   ACT 0: ReLU -> f16 out (stride outStride)
//   ACT 1: sigmoid -> f32 out (stride outStride)
// ---------------------------------------------------------------------------
template <int ACT>
__global__ __launch_bounds__(256)
void gemm_bias_act(const _Float16* __restrict__ A,
                   const _Float16* __restrict__ Bt,
                   const float* __restrict__ bias,
                   void* __restrict__ out,
                   int Kpad, int outStride) {
    const int lane  = threadIdx.x & 31;
    const int wave  = threadIdx.x >> 5;
    const int mBase = blockIdx.y * 256 + wave * 32;
    const int nBase = blockIdx.x * 64;

    const int l15 = lane & 15;
    const int kb  = (lane >> 4) << 3;     // 0 or 8 (K sub-block per half-wave)

    const _Float16* Arow0 = A  + (size_t)(mBase +      l15) * Kpad;
    const _Float16* Arow1 = A  + (size_t)(mBase + 16 + l15) * Kpad;
    const _Float16* Br0   = Bt + (size_t)(nBase +  0 + l15) * Kpad;
    const _Float16* Br1   = Bt + (size_t)(nBase + 16 + l15) * Kpad;
    const _Float16* Br2   = Bt + (size_t)(nBase + 32 + l15) * Kpad;
    const _Float16* Br3   = Bt + (size_t)(nBase + 48 + l15) * Kpad;

    v8f acc[2][4] = {{v8f{}, v8f{}, v8f{}, v8f{}},
                     {v8f{}, v8f{}, v8f{}, v8f{}}};

    for (int k0 = 0; k0 < Kpad; k0 += 32) {
        FragH a0, a1, b0, b1, b2, b3;
        a0.h[0] = *(const v8h*)(Arow0 + k0 + kb);
        a0.h[1] = *(const v8h*)(Arow0 + k0 + kb + 16);
        a1.h[0] = *(const v8h*)(Arow1 + k0 + kb);
        a1.h[1] = *(const v8h*)(Arow1 + k0 + kb + 16);
        b0.h[0] = *(const v8h*)(Br0   + k0 + kb);
        b0.h[1] = *(const v8h*)(Br0   + k0 + kb + 16);
        b1.h[0] = *(const v8h*)(Br1   + k0 + kb);
        b1.h[1] = *(const v8h*)(Br1   + k0 + kb + 16);
        b2.h[0] = *(const v8h*)(Br2   + k0 + kb);
        b2.h[1] = *(const v8h*)(Br2   + k0 + kb + 16);
        b3.h[0] = *(const v8h*)(Br3   + k0 + kb);
        b3.h[1] = *(const v8h*)(Br3   + k0 + kb + 16);

        acc[0][0] = __builtin_amdgcn_wmma_f32_16x16x32_f16(false, a0.v, false, b0.v,
                                                           (short)0, acc[0][0], false, false);
        acc[0][1] = __builtin_amdgcn_wmma_f32_16x16x32_f16(false, a0.v, false, b1.v,
                                                           (short)0, acc[0][1], false, false);
        acc[0][2] = __builtin_amdgcn_wmma_f32_16x16x32_f16(false, a0.v, false, b2.v,
                                                           (short)0, acc[0][2], false, false);
        acc[0][3] = __builtin_amdgcn_wmma_f32_16x16x32_f16(false, a0.v, false, b3.v,
                                                           (short)0, acc[0][3], false, false);
        acc[1][0] = __builtin_amdgcn_wmma_f32_16x16x32_f16(false, a1.v, false, b0.v,
                                                           (short)0, acc[1][0], false, false);
        acc[1][1] = __builtin_amdgcn_wmma_f32_16x16x32_f16(false, a1.v, false, b1.v,
                                                           (short)0, acc[1][1], false, false);
        acc[1][2] = __builtin_amdgcn_wmma_f32_16x16x32_f16(false, a1.v, false, b2.v,
                                                           (short)0, acc[1][2], false, false);
        acc[1][3] = __builtin_amdgcn_wmma_f32_16x16x32_f16(false, a1.v, false, b3.v,
                                                           (short)0, acc[1][3], false, false);
    }

    // Epilogue.  C/D layout: lane l15 = column, VGPR v = row v + 8*(lane>=16).
    const int mTop = mBase + ((lane >> 4) << 3);
    #pragma unroll
    for (int mt = 0; mt < 2; ++mt) {
        #pragma unroll
        for (int nt = 0; nt < 4; ++nt) {
            const int n = nBase + nt * 16 + l15;
            const float bv = bias[n];
            #pragma unroll
            for (int v = 0; v < 8; ++v) {
                const int m = mTop + mt * 16 + v;
                float val = acc[mt][nt][v] + bv;
                if (ACT == 0) {
                    val = fmaxf(val, 0.0f);
                    ((_Float16*)out)[(size_t)m * outStride + n] = (_Float16)val;
                } else {
                    val = 1.0f / (1.0f + __expf(-val));
                    ((float*)out)[(size_t)m * outStride + n] = val;
                }
            }
        }
    }
}

// ---------------------------------------------------------------------------
// Launch
// ---------------------------------------------------------------------------
extern "C" void kernel_launch(void* const* d_in, const int* in_sizes, int n_in,
                              void* d_out, int out_size, void* d_ws, size_t ws_size,
                              hipStream_t stream) {
    const float* x  = (const float*)d_in[0];   // (B, 12)
    const float* qp = (const float*)d_in[1];   // (4, 72)
    const float* W1 = (const float*)d_in[2];   // (256, 500)
    const float* b1 = (const float*)d_in[3];   // (500,)
    const float* W2 = (const float*)d_in[4];   // (500, 500)
    const float* b2 = (const float*)d_in[5];   // (500,)
    const float* W3 = (const float*)d_in[6];   // (500, 256)
    const float* b3 = (const float*)d_in[7];   // (256,)
    float* out = (float*)d_out;                // (B, 256) f32

    char* ws = (char*)d_ws;
    _Float16* feat = (_Float16*)ws;            ws += (size_t)BATCH * IMG  * 2;  // 4 MB
    _Float16* h1   = (_Float16*)ws;            ws += (size_t)BATCH * HPAD * 2;  // 8 MB
    _Float16* h2   = (_Float16*)ws;            ws += (size_t)BATCH * HPAD * 2;  // 8 MB
    _Float16* W1t  = (_Float16*)ws;            ws += (size_t)HPAD * IMG  * 2;   // 256 KB
    _Float16* W2t  = (_Float16*)ws;            ws += (size_t)HPAD * HPAD * 2;   // 512 KB
    _Float16* W3t  = (_Float16*)ws;            ws += (size_t)IMG  * HPAD * 2;   // 256 KB
    float*    b1p  = (float*)ws;               ws += HPAD * 4;
    float*    b2p  = (float*)ws;               ws += HPAD * 4;
    float*    b3p  = (float*)ws;               ws += IMG * 4;

    // --- weight / bias prep (f32 -> f16, transpose, zero-pad 500 -> 512) ---
    transpose_pad_f16<<<(IMG  * HPAD + 255) / 256, 256, 0, stream>>>(W1, W1t, IMG, 500, IMG,  HPAD);
    transpose_pad_f16<<<(HPAD * HPAD + 255) / 256, 256, 0, stream>>>(W2, W2t, 500, 500, HPAD, HPAD);
    transpose_pad_f16<<<(HPAD * IMG  + 255) / 256, 256, 0, stream>>>(W3, W3t, 500, IMG, HPAD, IMG);
    pad_bias<<<(HPAD + 255) / 256, 256, 0, stream>>>(b1, b1p, 500, HPAD);
    pad_bias<<<(HPAD + 255) / 256, 256, 0, stream>>>(b2, b2p, 500, HPAD);
    pad_bias<<<(IMG  + 255) / 256, 256, 0, stream>>>(b3, b3p, IMG, IMG);

    // --- quantum simulation: 1 wave per (gen, sample), 8 waves/block ---
    qsim_kernel<<<(NGEN * BATCH) / 8, 256, 0, stream>>>(x, qp, feat);

    // --- MLP with f16 WMMA, f32 accumulate ---
    // layer 1: (B,256) @ (256,512) -> h1 (B,512), ReLU
    gemm_bias_act<0><<<dim3(HPAD / 64, BATCH / 256), 256, 0, stream>>>(feat, W1t, b1p, h1, IMG,  HPAD);
    // layer 2: (B,512) @ (512,512) -> h2 (B,512), ReLU
    gemm_bias_act<0><<<dim3(HPAD / 64, BATCH / 256), 256, 0, stream>>>(h1,   W2t, b2p, h2, HPAD, HPAD);
    // layer 3: (B,512) @ (512,256) -> out (B,256), sigmoid
    gemm_bias_act<1><<<dim3(IMG  / 64, BATCH / 256), 256, 0, stream>>>(h2,   W3t, b3p, out, HPAD, IMG);
}
